// MultiHeadAttention_5677946765706
// MI455X (gfx1250) — matrix operand order repaired and checked by reading, not compiled, not run on hardware
//
#include <hip/hip_runtime.h>

// ---------------- CDNA5 WMMA types ----------------
typedef _Float16 v16h __attribute__((ext_vector_type(16)));
typedef _Float16 h2v  __attribute__((ext_vector_type(2)));
typedef _Float16 h4v  __attribute__((ext_vector_type(4)));
typedef float    v8f  __attribute__((ext_vector_type(8)));
typedef float    f4v  __attribute__((ext_vector_type(4)));

#define D_MODEL   512
#define NUM_HEADS 8
#define D_K       64
#define SEQ_N     2048
#define BATCH     4
#define M_ROWS    (BATCH * SEQ_N)   // 8192
#define NUM_BUCKETS 64

static __device__ __forceinline__ v8f wmma16(v16h a, v16h b, v8f c) {
  // D = A(16x32 f16) * B(32x16 f16) + C(16x16 f32)
  return __builtin_amdgcn_wmma_f32_16x16x32_f16(false, a, false, b, (short)0, c,
                                                false, false);
}

// A fragment (16x32, f16). rowbase = &A[row_of_this_lane][k0], row-major,
// contraction dim innermost. Lane = g*16 + m. Per ISA: pair j holds
// K = 2j+8g (j<4) or 16+2(j-4)+8g (j>=4)  -> half2 index (j<4?j:j+4)+4g.
// Two groups of 4 consecutive dwords -> 2x global_load_b128.
static __device__ __forceinline__ v16h load_frag_a(const _Float16* rowbase, int g) {
  const h2v* p = (const h2v*)rowbase;
  v16h a;
#pragma unroll
  for (int j = 0; j < 8; ++j) {
    h2v t = p[(j < 4 ? j : j + 4) + 4 * g];
    a[2 * j] = t[0];
    a[2 * j + 1] = t[1];
  }
  return a;
}

// B fragment (32x16, f16) sourced from B^T stored row-major:
// colbase = &Bt[n_of_this_lane][k0]. Lane = g*16 + n. Pair j holds
// K = 2j+8g (j<4) or 16+2(j-4)+8g (j>=4) -> half2 index (j<4?j:j+4)+4g.
static __device__ __forceinline__ v16h load_frag_b(const _Float16* colbase, int g) {
  const h2v* p = (const h2v*)colbase;
  v16h b;
#pragma unroll
  for (int j = 0; j < 8; ++j) {
    h2v t = p[(j < 4 ? j : j + 4) + 4 * g];
    b[2 * j] = t[0];
    b[2 * j + 1] = t[1];
  }
  return b;
}

// Reductions across the 16 lanes holding one WMMA output row (xor masks <=8
// stay inside each 16-lane half of the wave32).
static __device__ __forceinline__ float redmax16(float v) {
  v = fmaxf(v, __shfl_xor(v, 1, 32));
  v = fmaxf(v, __shfl_xor(v, 2, 32));
  v = fmaxf(v, __shfl_xor(v, 4, 32));
  v = fmaxf(v, __shfl_xor(v, 8, 32));
  return v;
}
static __device__ __forceinline__ float redsum16(float v) {
  v += __shfl_xor(v, 1, 32);
  v += __shfl_xor(v, 2, 32);
  v += __shfl_xor(v, 4, 32);
  v += __shfl_xor(v, 8, 32);
  return v;
}

// Raw v_sqrt_f32 (no IEEE range-fixup ladder): plenty for 0.1-wide bucketing.
static __device__ __forceinline__ float fast_sqrt(float x) {
  return __builtin_amdgcn_sqrtf(x);
}

// ---------------- fp32 -> fp16 convert (vectorized, b128 loads) ------------
__global__ void f32_to_f16_kernel(const float* __restrict__ s,
                                  _Float16* __restrict__ d, int n4) {
  const f4v* __restrict__ sv = (const f4v*)s;
  h4v* __restrict__ dv = (h4v*)d;
  for (int i = blockIdx.x * blockDim.x + threadIdx.x; i < n4;
       i += gridDim.x * blockDim.x)
    dv[i] = __builtin_convertvector(sv[i], h4v);
}

// ---------------- NT GEMM:  Out = A(f16)[M,512] @ Bt(f16)[512,512]^T + bias --
// MODE is a compile-time template parameter -> straight-line epilogue.
// MODE 0: Q projection  -> f16 [B,H,N,dk], pre-scaled by 1/sqrt(dk)
// MODE 1: K projection  -> f16 [B,H,N,dk]
// MODE 2: V projection  -> f16 [B,H,dk,N]   (transposed for PV B-fragments)
// MODE 3: out projection-> f32 [M,512] (d_out)
template <int MODE>
__global__ __launch_bounds__(256) void wmma_gemm_nt(
    const _Float16* __restrict__ A, const _Float16* __restrict__ Bt,
    const float* __restrict__ bias, void* __restrict__ Out) {
  const int wave = threadIdx.x >> 5;
  const int lane = threadIdx.x & 31;
  const int g = lane >> 4;
  const int ln = lane & 15;
  const int m0 = (blockIdx.x * 8 + wave) * 16;
  const int n0 = blockIdx.y * 64;

  v8f zf = {0.f, 0.f, 0.f, 0.f, 0.f, 0.f, 0.f, 0.f};
  v8f acc[4] = {zf, zf, zf, zf};

  const _Float16* arow = A + (size_t)(m0 + ln) * D_MODEL;
  const _Float16* brow = Bt + (size_t)(n0 + ln) * D_MODEL;

  for (int kk = 0; kk < D_MODEL; kk += 32) {
    v16h a = load_frag_a(arow + kk, g);
#pragma unroll
    for (int t = 0; t < 4; ++t) {
      v16h b = load_frag_b(brow + (size_t)(16 * t) * D_MODEL + kk, g);
      acc[t] = wmma16(a, b, acc[t]);
    }
  }

#pragma unroll
  for (int t = 0; t < 4; ++t) {
    const int o = n0 + 16 * t + ln;
    const float bv = bias[o];
#pragma unroll
    for (int r = 0; r < 8; ++r) {
      const int row = m0 + 8 * g + r;  // D layout: m = r + 8*(lane/16)
      const float v = acc[t][r] + bv;
      if (MODE == 3) {
        ((float*)Out)[(size_t)row * D_MODEL + o] = v;
      } else {
        const int bb = row >> 11, ns = row & (SEQ_N - 1);
        const int hh = o >> 6, d = o & (D_K - 1);
        _Float16* o16 = (_Float16*)Out;
        if (MODE == 0) {
          o16[(((size_t)bb * NUM_HEADS + hh) * SEQ_N + ns) * D_K + d] =
              (_Float16)(v * 0.125f);  // fold 1/sqrt(dk) into Q
        } else if (MODE == 1) {
          o16[(((size_t)bb * NUM_HEADS + hh) * SEQ_N + ns) * D_K + d] =
              (_Float16)v;
        } else {  // MODE == 2: V transposed
          o16[(((size_t)bb * NUM_HEADS + hh) * D_K + d) * SEQ_N + ns] =
              (_Float16)v;
        }
      }
    }
  }
}

// ---------------- Flash attention with distance-bucket bias ----------------
// Block = 256 threads = 8 waves; block covers 128 query rows of one (b,h).
// Each wave owns a 16-row Q tile and streams 32-key blocks; scores are never
// materialized (naive path would write+read a 512MB [B,H,N,N] tensor).
__global__ __launch_bounds__(256) void wmma_flash_attn(
    const _Float16* __restrict__ Qm, const _Float16* __restrict__ Km,
    const _Float16* __restrict__ Vt, const float* __restrict__ pos,
    const float* __restrict__ relpos, _Float16* __restrict__ ctx) {
  __shared__ _Float16 ldsP[8 * 16 * 32];  // per-wave P staging (D->A relayout)
  __shared__ float ldsBias[NUM_BUCKETS];

  const int b = blockIdx.z, h = blockIdx.y;
  const int tid = threadIdx.x;
  const int wave = tid >> 5, lane = tid & 31;
  const int g = lane >> 4, ln = lane & 15;

  if (tid < NUM_BUCKETS) ldsBias[tid] = relpos[h * NUM_BUCKETS + tid];
  __syncthreads();

  const int m0q = blockIdx.x * 128 + wave * 16;  // query row within sequence
  const size_t bh = (size_t)b * NUM_HEADS + h;
  const _Float16* Qb = Qm + bh * SEQ_N * D_K;
  const _Float16* Kb = Km + bh * SEQ_N * D_K;
  const _Float16* Vb = Vt + bh * D_K * SEQ_N;
  const float* posB = pos + (size_t)b * SEQ_N * 3;

  // Q fragments held for the whole key loop (dk=64 -> two K=32 chunks).
  const v16h qa0 = load_frag_a(Qb + (size_t)(m0q + ln) * D_K, g);
  const v16h qa1 = load_frag_a(Qb + (size_t)(m0q + ln) * D_K + 32, g);

  // Query positions for this lane's 8 output rows (m = 8g + r).
  float qx[8], qy[8], qz[8];
#pragma unroll
  for (int r = 0; r < 8; ++r) {
    const int qr = m0q + 8 * g + r;
    qx[r] = posB[qr * 3 + 0];
    qy[r] = posB[qr * 3 + 1];
    qz[r] = posB[qr * 3 + 2];
  }

  float mi[8], li[8];
#pragma unroll
  for (int r = 0; r < 8; ++r) { mi[r] = -3.0e38f; li[r] = 0.f; }
  v8f zf = {0.f, 0.f, 0.f, 0.f, 0.f, 0.f, 0.f, 0.f};
  v8f acc[4] = {zf, zf, zf, zf};

  _Float16* myP = ldsP + wave * (16 * 32);  // wave-private, DS ops in-order

  for (int j0 = 0; j0 < SEQ_N; j0 += 32) {
    if (j0 + 32 < SEQ_N) {  // hint next tiles (global_prefetch_b8)
      __builtin_prefetch(Kb + (size_t)(j0 + 32 + ln) * D_K, 0, 1);
      __builtin_prefetch(Vb + (size_t)ln * SEQ_N + j0 + 32, 0, 1);
    }

    // ---- S = (Q/sqrt(dk)) @ K^T for 32 keys: two 16x16 tiles ----
    v8f s0 = zf, s1 = zf;
    {
      const _Float16* k0 = Kb + (size_t)(j0 + ln) * D_K;
      s0 = wmma16(qa0, load_frag_b(k0, g), s0);
      s0 = wmma16(qa1, load_frag_b(k0 + 32, g), s0);
      const _Float16* k1 = Kb + (size_t)(j0 + 16 + ln) * D_K;
      s1 = wmma16(qa0, load_frag_b(k1, g), s1);
      s1 = wmma16(qa1, load_frag_b(k1 + 32, g), s1);
    }

    // ---- relative-position bucket bias, computed on the fly ----
    const int kj0 = j0 + ln, kj1 = j0 + 16 + ln;  // this lane's key columns
    const float k0x = posB[kj0 * 3], k0y = posB[kj0 * 3 + 1], k0z = posB[kj0 * 3 + 2];
    const float k1x = posB[kj1 * 3], k1y = posB[kj1 * 3 + 1], k1z = posB[kj1 * 3 + 2];
#pragma unroll
    for (int r = 0; r < 8; ++r) {
      float dx = qx[r] - k0x, dy = qy[r] - k0y, dz = qz[r] - k0z;
      int i0 = (int)(fast_sqrt(dx * dx + dy * dy + dz * dz) * 10.0f);
      i0 = i0 > 63 ? 63 : i0;
      s0[r] += ldsBias[i0];
      dx = qx[r] - k1x; dy = qy[r] - k1y; dz = qz[r] - k1z;
      int i1 = (int)(fast_sqrt(dx * dx + dy * dy + dz * dz) * 10.0f);
      i1 = i1 > 63 ? 63 : i1;
      s1[r] += ldsBias[i1];
    }

    // ---- online softmax update + stage P into LDS (A-matrix layout src) ----
#pragma unroll
    for (int r = 0; r < 8; ++r) {
      float bm = redmax16(fmaxf(s0[r], s1[r]));
      float mnew = fmaxf(mi[r], bm);
      float corr = __expf(mi[r] - mnew);
      float p0 = __expf(s0[r] - mnew);
      float p1 = __expf(s1[r] - mnew);
      float rs = redsum16(p0 + p1);
      li[r] = li[r] * corr + rs;
      mi[r] = mnew;
      acc[0][r] *= corr; acc[1][r] *= corr;
      acc[2][r] *= corr; acc[3][r] *= corr;
      myP[(8 * g + r) * 32 + ln] = (_Float16)p0;       // row m, cols 0..15
      myP[(8 * g + r) * 32 + 16 + ln] = (_Float16)p1;  // row m, cols 16..31
    }

    // ---- O += P(16x32) @ V(32x64) ----
    const v16h pa = load_frag_a(myP + (size_t)ln * 32, g);
#pragma unroll
    for (int t = 0; t < 4; ++t) {
      v16h vb = load_frag_b(Vb + (size_t)(16 * t + ln) * SEQ_N + j0, g);
      acc[t] = wmma16(pa, vb, acc[t]);
    }
  }

  // ---- normalize and write context (f16, [B*N, 512] row-major) ----
#pragma unroll
  for (int t = 0; t < 4; ++t) {
#pragma unroll
    for (int r = 0; r < 8; ++r) {
      const size_t row = (size_t)b * SEQ_N + m0q + 8 * g + r;
      ctx[row * D_MODEL + h * D_K + 16 * t + ln] =
          (_Float16)(acc[t][r] / li[r]);
    }
  }
}

// ---------------- host-side orchestration ----------------
extern "C" void kernel_launch(void* const* d_in, const int* in_sizes, int n_in,
                              void* d_out, int out_size, void* d_ws,
                              size_t ws_size, hipStream_t stream) {
  (void)in_sizes; (void)n_in; (void)out_size; (void)ws_size;
  const float* query = (const float*)d_in[0];
  const float* key_  = (const float*)d_in[1];
  const float* value = (const float*)d_in[2];
  const float* positions = (const float*)d_in[3];
  const float* wq = (const float*)d_in[4];
  const float* bq = (const float*)d_in[5];
  const float* wk = (const float*)d_in[6];
  const float* bk_ = (const float*)d_in[7];
  const float* wv = (const float*)d_in[8];
  const float* bv = (const float*)d_in[9];
  const float* wo = (const float*)d_in[10];
  const float* bo = (const float*)d_in[11];
  const float* relpos = (const float*)d_in[12];

  char* ws = (char*)d_ws;
  const size_t ACT = (size_t)M_ROWS * D_MODEL * sizeof(_Float16);  // 8 MiB
  const size_t WTS = (size_t)D_MODEL * D_MODEL * sizeof(_Float16); // 512 KiB
  size_t off = 0;
  _Float16* qh   = (_Float16*)(ws + off); off += ACT;  // later reused as ctx
  _Float16* kh   = (_Float16*)(ws + off); off += ACT;
  _Float16* vh   = (_Float16*)(ws + off); off += ACT;
  _Float16* Qws  = (_Float16*)(ws + off); off += ACT;  // [B,H,N,dk]
  _Float16* Kws  = (_Float16*)(ws + off); off += ACT;  // [B,H,N,dk]
  _Float16* Vtws = (_Float16*)(ws + off); off += ACT;  // [B,H,dk,N]
  _Float16* wqh  = (_Float16*)(ws + off); off += WTS;
  _Float16* wkh  = (_Float16*)(ws + off); off += WTS;
  _Float16* wvh  = (_Float16*)(ws + off); off += WTS;
  _Float16* woh  = (_Float16*)(ws + off); off += WTS;

  const int NACT4 = (M_ROWS * D_MODEL) / 4;
  const int NW4 = (D_MODEL * D_MODEL) / 4;
  f32_to_f16_kernel<<<512, 256, 0, stream>>>(query, qh, NACT4);
  f32_to_f16_kernel<<<512, 256, 0, stream>>>(key_, kh, NACT4);
  f32_to_f16_kernel<<<512, 256, 0, stream>>>(value, vh, NACT4);
  f32_to_f16_kernel<<<64, 256, 0, stream>>>(wq, wqh, NW4);
  f32_to_f16_kernel<<<64, 256, 0, stream>>>(wk, wkh, NW4);
  f32_to_f16_kernel<<<64, 256, 0, stream>>>(wv, wvh, NW4);
  f32_to_f16_kernel<<<64, 256, 0, stream>>>(wo, woh, NW4);

  dim3 gg(M_ROWS / 16 / 8, D_MODEL / 64);  // (64, 8)
  wmma_gemm_nt<0><<<gg, 256, 0, stream>>>(qh, wqh, bq, (void*)Qws);
  wmma_gemm_nt<1><<<gg, 256, 0, stream>>>(kh, wkh, bk_, (void*)Kws);
  wmma_gemm_nt<2><<<gg, 256, 0, stream>>>(vh, wvh, bv, (void*)Vtws);

  _Float16* ctx = qh;  // qh is dead after the Q projection; reuse for context
  wmma_flash_attn<<<dim3(SEQ_N / 128, NUM_HEADS, BATCH), 256, 0, stream>>>(
      Qws, Kws, Vtws, positions, relpos, ctx);

  wmma_gemm_nt<3><<<gg, 256, 0, stream>>>(ctx, woh, bo, d_out);
}
